// G1_3556232921667
// MI455X (gfx1250) — matrix operand order repaired and checked by reading, not compiled, not run on hardware
//
#include <hip/hip_runtime.h>

typedef __attribute__((ext_vector_type(16))) __bf16 v16bf;
typedef __attribute__((ext_vector_type(8)))  float  v8f;
typedef __attribute__((ext_vector_type(4)))  int    v4i;

#define NKT      20      // 640 / 32 k-tiles
#define KSTEP    32
#define NNT      16      // 256 / 16 n-tiles
#define BLOCK_M  128
#define N_EDGES  320000

#if __has_builtin(__builtin_amdgcn_global_load_async_to_lds_b128) && \
    __has_builtin(__builtin_amdgcn_s_wait_asynccnt)
#define USE_ASYNC 1
#else
#define USE_ASYNC 0
#endif

// float -> bf16 via compiler fptrunc (RNE); pack two into a dword
static __device__ __forceinline__ unsigned pk2(float a, float b) {
  union { __bf16 h[2]; unsigned u; } x;
  x.h[0] = (__bf16)a;
  x.h[1] = (__bf16)b;
  return x.u;
}

// ---------------------------------------------------------------------------
// Pack [w1;w2;w3] (640x256 fp32) into bf16 B-fragments in d_ws (320 KB).
// Fragment (kt,nt) = 256 dwords: lane l holds N = nt*16+(l&15),
// K = kt*32 + (l>>4)*16 + j (j=0..15), pairs per dword.
// Bank swizzle: lanes with bit3 set swap their two 16B chunks so a 16-lane
// ds_load_b128 phase covers all 64 LDS banks.
// ---------------------------------------------------------------------------
__global__ void pack_w_kernel(const float* __restrict__ w1,
                              const float* __restrict__ w2,
                              const float* __restrict__ w3,
                              unsigned* __restrict__ wsb) {
  const int frag = blockIdx.x;          // kt*16 + nt
  const int kt   = frag >> 4;
  const int nt   = frag & 15;
  const int lane = threadIdx.x;         // 0..31
  const int n    = nt * 16 + (lane & 15);
  const int k0   = kt * 32 + ((lane >> 4) << 4);

  float vals[16];
#pragma unroll
  for (int j = 0; j < 16; ++j) {
    const int k = k0 + j;
    const float* w; int kk;
    if (k < 256)      { w = w1; kk = k;       }
    else if (k < 384) { w = w2; kk = k - 256; }
    else              { w = w3; kk = k - 384; }
    vals[j] = w[(size_t)kk * 256 + n];
  }

  unsigned u[8];
#pragma unroll
  for (int i = 0; i < 8; ++i)
    u[i] = pk2(vals[2 * i], vals[2 * i + 1]);

  const int s    = (lane >> 3) & 1;
  const int off0 = lane * 8 + s * 4;          // dwords, chunk holding j=0..7
  const int off1 = lane * 8 + (s ^ 1) * 4;    // dwords, chunk holding j=8..15
  unsigned* fragp = wsb + (size_t)frag * 256;
  uint4 q0; q0.x = u[0]; q0.y = u[1]; q0.z = u[2]; q0.w = u[3];
  uint4 q1; q1.x = u[4]; q1.y = u[5]; q1.z = u[6]; q1.w = u[7];
  *(uint4*)(fragp + off0) = q0;
  *(uint4*)(fragp + off1) = q1;
}

// ---------------------------------------------------------------------------
// Stage one 16 KB B k-tile: global (pre-packed bf16) -> LDS.
// Async path: 4 x global_load_async_to_lds_b128 per thread, no data VGPRs.
// ---------------------------------------------------------------------------
static __device__ __forceinline__ void stage_tile(const unsigned* __restrict__ gsrc,
                                                  unsigned* ldst, int tid) {
#if USE_ASYNC
  typedef __attribute__((address_space(1))) v4i gv4i;
  typedef __attribute__((address_space(3))) v4i lv4i;
  gv4i* g = (gv4i*)(gsrc + tid * 4);
  lv4i* l = (lv4i*)(ldst + tid * 4);
  __builtin_amdgcn_global_load_async_to_lds_b128(g, l, 0,     0);
  __builtin_amdgcn_global_load_async_to_lds_b128(g, l, 4096,  0);
  __builtin_amdgcn_global_load_async_to_lds_b128(g, l, 8192,  0);
  __builtin_amdgcn_global_load_async_to_lds_b128(g, l, 12288, 0);
#else
  uint4 tmp[4];
#pragma unroll
  for (int i = 0; i < 4; ++i) tmp[i] = ((const uint4*)gsrc)[i * 256 + tid];
#pragma unroll
  for (int i = 0; i < 4; ++i) ((uint4*)ldst)[i * 256 + tid] = tmp[i];
#endif
}

static __device__ __forceinline__ void stage_wait() {
#if USE_ASYNC
  __builtin_amdgcn_s_wait_asynccnt(0);
#endif
}

// ---------------------------------------------------------------------------
// Fused GEMM + bias + relu.
// Block: 256 threads = 8 waves; 128 rows x 256 cols per block.
// Wave w: rows [blk*128 + w*16, +16), all 16 n-tiles -> 16 x v8f accumulators.
// B double-buffered through LDS via async copy; A from global fp32 -> bf16.
// ---------------------------------------------------------------------------
__global__ __launch_bounds__(256)
void gemm_bias_relu_kernel(const float* __restrict__ f_src,
                           const float* __restrict__ f,
                           const float* __restrict__ sum_msg,
                           const float* __restrict__ bias,
                           const unsigned* __restrict__ wsb,
                           float* __restrict__ out) {
  __shared__ __align__(16) unsigned ldsB[2][4096];   // 2 x 16 KB

  const int tid     = threadIdx.x;
  const int lane    = tid & 31;
  const int wave    = tid >> 5;
  const int l15     = lane & 15;
  const int hi      = lane >> 4;                     // 0 or 1
  const int rowbase = blockIdx.x * BLOCK_M + wave * 16;
  const int arow    = rowbase + l15;

  // per-lane swizzled fragment offsets (dwords) — must match pack_w_kernel
  const int s    = (lane >> 3) & 1;
  const int off0 = lane * 8 + s * 4;
  const int off1 = lane * 8 + (s ^ 1) * 4;

  v8f acc[NNT];
#pragma unroll
  for (int t = 0; t < NNT; ++t)
    acc[t] = (v8f){0.f, 0.f, 0.f, 0.f, 0.f, 0.f, 0.f, 0.f};

  // stage k-tile 0
  stage_tile(wsb, ldsB[0], tid);
  stage_wait();
  __syncthreads();

  for (int kt = 0; kt < NKT; ++kt) {
    const int cur = kt & 1;

    // start async staging of the next B k-tile into the other buffer
    if (kt + 1 < NKT)
      stage_tile(wsb + (size_t)(kt + 1) * 4096, ldsB[1 - cur], tid);

    // ---- A fragment: 16x32 bf16 per wave ----
    const int kbase = kt * KSTEP;
    const float* src; int stride, kofs;
    if (kbase < 256)      { src = f_src;   stride = 256; kofs = kbase;       }
    else if (kbase < 384) { src = f;       stride = 128; kofs = kbase - 256; }
    else                  { src = sum_msg; stride = 256; kofs = kbase - 384; }

    const float* ap = src + (size_t)arow * stride + kofs + hi * 16;
    float4 a0 = *(const float4*)(ap);
    float4 a1 = *(const float4*)(ap + 4);
    float4 a2 = *(const float4*)(ap + 8);
    float4 a3 = *(const float4*)(ap + 12);

    union { v16bf v; unsigned u[8]; } af;
    af.u[0] = pk2(a0.x, a0.y);  af.u[1] = pk2(a0.z, a0.w);
    af.u[2] = pk2(a1.x, a1.y);  af.u[3] = pk2(a1.z, a1.w);
    af.u[4] = pk2(a2.x, a2.y);  af.u[5] = pk2(a2.z, a2.w);
    af.u[6] = pk2(a3.x, a3.y);  af.u[7] = pk2(a3.z, a3.w);

    // ---- 16 WMMAs against LDS-resident B fragments ----
    const unsigned* fb = ldsB[cur];
    const uint4* p0 = (const uint4*)(fb + off0);
    const uint4* p1 = (const uint4*)(fb + off1);
#pragma unroll
    for (int nt = 0; nt < NNT; ++nt) {
      union { v16bf v; uint4 q[2]; } bq;
      bq.q[0] = p0[nt * 64];          // ds_load_b128, imm offset nt*1024
      bq.q[1] = p1[nt * 64];
      acc[nt] = __builtin_amdgcn_wmma_f32_16x16x32_bf16(
          /*neg_a=*/false, af.v, /*neg_b=*/false, bq.v,
          /*c_mod=*/(short)0, acc[nt], /*reuse_a=*/false, /*reuse_b=*/false);
    }

    stage_wait();       // next tile fully in LDS (this wave's copies)
    __syncthreads();    // all waves done reading cur + done staging next
  }

  // ---- epilogue: bias + relu, coalesced stores ----
#pragma unroll
  for (int nt = 0; nt < NNT; ++nt) {
    const float bv = bias[nt * 16 + l15];
#pragma unroll
    for (int v = 0; v < 8; ++v) {
      float x = acc[nt][v] + bv;
      x = x > 0.f ? x : 0.f;
      out[(size_t)(rowbase + v + 8 * hi) * 256 + nt * 16 + l15] = x;
    }
  }
}

// ---------------------------------------------------------------------------
extern "C" void kernel_launch(void* const* d_in, const int* in_sizes, int n_in,
                              void* d_out, int out_size, void* d_ws, size_t ws_size,
                              hipStream_t stream) {
  (void)in_sizes; (void)n_in; (void)out_size; (void)ws_size;
  const float* f_src   = (const float*)d_in[0];
  const float* f       = (const float*)d_in[1];
  const float* sum_msg = (const float*)d_in[2];
  const float* w1      = (const float*)d_in[3];
  const float* w2      = (const float*)d_in[4];
  const float* w3      = (const float*)d_in[5];
  const float* bias    = (const float*)d_in[6];
  unsigned* wpack      = (unsigned*)d_ws;          // 320 KB of scratch used

  pack_w_kernel<<<NKT * NNT, 32, 0, stream>>>(w1, w2, w3, wpack);
  gemm_bias_relu_kernel<<<N_EDGES / BLOCK_M, 256, 0, stream>>>(
      f_src, f, sum_msg, bias, wpack, (float*)d_out);
}
